// RAU_9268539424851
// MI455X (gfx1250) — matrix operand order repaired and checked by reading, not compiled, not run on hardware
//
#include <hip/hip_runtime.h>
#include <hip/hip_bf16.h>
#include <math.h>

// Problem constants
#define B_ 128
#define S_ 256
#define I_ 512
#define H_ 1024
#define G_ (3 * H_)

typedef __attribute__((ext_vector_type(16))) __bf16 bf16x16;
typedef __attribute__((ext_vector_type(8)))  float  f32x8;

union ABfrag { bf16x16 v; uint4 q[2]; };

__device__ __forceinline__ unsigned short f32_to_bf16_rne(float f) {
    unsigned int u = __float_as_uint(f);
    unsigned int r = u + 0x7FFFu + ((u >> 16) & 1u);
    return (unsigned short)(r >> 16);
}

// ---------------------------------------------------------------------------
// fp32 -> bf16 conversion (grid-stride)
// ---------------------------------------------------------------------------
__global__ void cvt_bf16_kernel(const float* __restrict__ src,
                                unsigned short* __restrict__ dst, int n) {
    for (int i = blockIdx.x * blockDim.x + threadIdx.x; i < n;
         i += gridDim.x * blockDim.x)
        dst[i] = f32_to_bf16_rne(src[i]);
}

// Zero the initial hidden state (fp32 + bf16 copies)
__global__ void zero_h_kernel(float* __restrict__ h,
                              unsigned short* __restrict__ hb, int n) {
    for (int i = blockIdx.x * blockDim.x + threadIdx.x; i < n;
         i += gridDim.x * blockDim.x) {
        h[i] = 0.0f;
        hb[i] = 0;
    }
}

// ---------------------------------------------------------------------------
// sx[b,s] = x[b,s,:] . W_att[0, :I]   (one wave per row, shfl reduction)
// ---------------------------------------------------------------------------
__global__ void sx_kernel(const float* __restrict__ x,
                          const float* __restrict__ Watt,
                          float* __restrict__ sx) {
    int row  = blockIdx.x * 8 + (threadIdx.x >> 5);
    int lane = threadIdx.x & 31;
    const float* xp = x + (size_t)row * I_;
    float s = 0.0f;
    for (int k = lane; k < I_; k += 32) s += xp[k] * Watt[k];
    for (int off = 16; off; off >>= 1) s += __shfl_down(s, off, 32);
    if (lane == 0) sx[row] = s;
}

// ---------------------------------------------------------------------------
// Wave32 bf16 WMMA GEMM:  C[M,N] = A[M,K] * W[N,K]^T   (fp32 accumulate)
// Block = 8 waves (2x4). Wave tile = 32 rows x (16*CT) cols, K-step 32.
// Block tile = 64 x (64*CT). Per-lane fragment loads follow the CDNA5
// 16-bit A (K 0..7 / 16..23 per half-wave) and B (same K 0..7 / 16..23 per
// half-wave) VGPR layouts, so no LDS shuffle is required.
// ---------------------------------------------------------------------------
template <int CT>
__global__ void __launch_bounds__(256)
gemm_bf16_nt(const unsigned short* __restrict__ A,
             const unsigned short* __restrict__ W,
             float* __restrict__ C, int M, int N, int K) {
    const int lane = threadIdx.x & 31;
    const int wave = threadIdx.x >> 5;
    const int wr   = wave >> 2;    // 0..1
    const int wc   = wave & 3;     // 0..3
    const int row0 = blockIdx.x * 64 + wr * 32;
    const int col0 = blockIdx.y * (64 * CT) + wc * (16 * CT);

    const int arow = lane & 15;
    const int ak8  = (lane >> 4) << 3;   // A: +8 K for upper half-wave
    const int bcol = lane & 15;
    const int bk16 = (lane >> 4) << 3;   // B: +8 K for upper half-wave

    const unsigned short* A0 = A + (size_t)(row0 + arow) * K + ak8;
    const unsigned short* A1 = A0 + (size_t)16 * K;
    const unsigned short* Bp[CT];
    #pragma unroll
    for (int j = 0; j < CT; ++j)
        Bp[j] = W + (size_t)(col0 + 16 * j + bcol) * K + bk16;

    f32x8 acc0[CT], acc1[CT];
    #pragma unroll
    for (int j = 0; j < CT; ++j) { acc0[j] = (f32x8){}; acc1[j] = (f32x8){}; }

    for (int k0 = 0; k0 < K; k0 += 32) {
        ABfrag a0, a1;
        a0.q[0] = *(const uint4*)(A0 + k0);       // K = k0    .. k0+7  (+ak8)
        a0.q[1] = *(const uint4*)(A0 + k0 + 16);  // K = k0+16 .. k0+23 (+ak8)
        a1.q[0] = *(const uint4*)(A1 + k0);
        a1.q[1] = *(const uint4*)(A1 + k0 + 16);
        ABfrag b[CT];
        #pragma unroll
        for (int j = 0; j < CT; ++j) {
            b[j].q[0] = *(const uint4*)(Bp[j] + k0);
            b[j].q[1] = *(const uint4*)(Bp[j] + k0 + 16);
        }

        if (k0 + 128 < K) {  // WGP-scope prefetch (locality 3)
            __builtin_prefetch((const void*)(A0 + k0 + 128), 0, 3);
            __builtin_prefetch((const void*)(A1 + k0 + 128), 0, 3);
            #pragma unroll
            for (int j = 0; j < CT; ++j)
                __builtin_prefetch((const void*)(Bp[j] + k0 + 128), 0, 3);
        }

        #pragma unroll
        for (int j = 0; j < CT; ++j) {
            acc0[j] = __builtin_amdgcn_wmma_f32_16x16x32_bf16(
                false, a0.v, false, b[j].v, (short)0, acc0[j], false, false);
            acc1[j] = __builtin_amdgcn_wmma_f32_16x16x32_bf16(
                false, a1.v, false, b[j].v, (short)0, acc1[j], false, false);
        }
    }

    // C/D layout: VGPR v -> row (v + 8*halfwave), col = lane&15
    const int crow = (lane >> 4) << 3;
    const int ccol = lane & 15;
    #pragma unroll
    for (int j = 0; j < CT; ++j) {
        float* p0 = C + (size_t)(row0 + crow) * N + (col0 + 16 * j + ccol);
        float* p1 = p0 + (size_t)16 * N;
        #pragma unroll
        for (int v = 0; v < 8; ++v) p0[(size_t)v * N] = acc0[j][v];
        #pragma unroll
        for (int v = 0; v < 8; ++v) p1[(size_t)v * N] = acc1[j][v];
    }
}

// ---------------------------------------------------------------------------
// Per-step attention softmax over the batch dim (128 scores, one block)
// score[b] = sx[b,t] + h[b,:].W_att[0,I:] + b_att ; att = softmax_b(score)
// ---------------------------------------------------------------------------
__global__ void att_kernel(const float* __restrict__ h,
                           const float* __restrict__ Watt,
                           const float* __restrict__ b_att,
                           const float* __restrict__ sx, int t,
                           float* __restrict__ att) {
    __shared__ float sm[B_];
    const int b = threadIdx.x;  // blockDim.x == 128
    const float* wh = Watt + I_;
    const float* hp = h + (size_t)b * H_;
    float s = 0.0f;
    #pragma unroll 4
    for (int k = 0; k < H_; ++k) s += hp[k] * wh[k];
    s += sx[b * S_ + t] + b_att[0];
    sm[b] = s;
    __syncthreads();
    for (int off = 64; off; off >>= 1) {
        if (b < off) sm[b] = fmaxf(sm[b], sm[b + off]);
        __syncthreads();
    }
    const float m = sm[0];
    __syncthreads();
    const float e = __expf(s - m);
    sm[b] = e;
    __syncthreads();
    for (int off = 64; off; off >>= 1) {
        if (b < off) sm[b] += sm[b + off];
        __syncthreads();
    }
    att[b] = e / sm[0];
}

// ---------------------------------------------------------------------------
// Fused GRU gate update:
//   gi = att[b]*GI[b,t,:] + b_ih ; gh' = gh + b_hh
//   r = sig(gi_r+gh_r), z = sig(gi_z+gh_z), n = tanh(gi_n + r*gh_n)
//   h' = (1-z)*n + z*h ; writes h' (fp32 + bf16) and outputs[b,t,:]
// ---------------------------------------------------------------------------
__global__ void combine_kernel(const float* __restrict__ GI,
                               const float* __restrict__ gh,
                               const float* __restrict__ att,
                               const float* __restrict__ b_ih,
                               const float* __restrict__ b_hh,
                               const float* __restrict__ h_prev,
                               float* __restrict__ h_new,
                               unsigned short* __restrict__ hbf_new,
                               float* __restrict__ out,
                               float* __restrict__ h_last, int t) {
    const int idx = blockIdx.x * blockDim.x + threadIdx.x;
    if (idx >= B_ * H_) return;
    const int b = idx >> 10;
    const int j = idx & (H_ - 1);
    const float a = att[b];
    const float* gi  = GI + (size_t)(b * S_ + t) * G_;
    const float* ghp = gh + (size_t)b * G_;
    const float ir = a * gi[j]            + b_ih[j];
    const float iz = a * gi[H_ + j]       + b_ih[H_ + j];
    const float in = a * gi[2 * H_ + j]   + b_ih[2 * H_ + j];
    const float hr = ghp[j]               + b_hh[j];
    const float hz = ghp[H_ + j]          + b_hh[H_ + j];
    const float hn = ghp[2 * H_ + j]      + b_hh[2 * H_ + j];
    const float r = 1.0f / (1.0f + __expf(-(ir + hr)));
    const float z = 1.0f / (1.0f + __expf(-(iz + hz)));
    const float n = tanhf(in + r * hn);
    const float hv = (1.0f - z) * n + z * h_prev[idx];
    h_new[idx]  = hv;
    hbf_new[idx] = f32_to_bf16_rne(hv);
    out[((size_t)b * S_ + t) * H_ + j] = hv;
    if (h_last) h_last[idx] = hv;
}

// ---------------------------------------------------------------------------
// Launcher
// ---------------------------------------------------------------------------
extern "C" void kernel_launch(void* const* d_in, const int* in_sizes, int n_in,
                              void* d_out, int out_size, void* d_ws, size_t ws_size,
                              hipStream_t stream) {
    (void)in_sizes; (void)n_in; (void)out_size; (void)ws_size;

    const float* x    = (const float*)d_in[0];
    const float* Watt = (const float*)d_in[1];
    const float* batt = (const float*)d_in[2];
    const float* Wih  = (const float*)d_in[3];
    const float* Whh  = (const float*)d_in[4];
    const float* bih  = (const float*)d_in[5];
    const float* bhh  = (const float*)d_in[6];

    char* ws = (char*)d_ws;
    size_t off = 0;
    auto take = [&](size_t bytes) -> char* {
        char* p = ws + off;
        off += (bytes + 255) & ~size_t(255);
        return p;
    };
    unsigned short* xbf  = (unsigned short*)take((size_t)B_ * S_ * I_ * 2);
    unsigned short* wihb = (unsigned short*)take((size_t)G_ * I_ * 2);
    unsigned short* whhb = (unsigned short*)take((size_t)G_ * H_ * 2);
    float* GI   = (float*)take((size_t)B_ * S_ * G_ * 4);
    float* sx   = (float*)take((size_t)B_ * S_ * 4);
    float* attb = (float*)take((size_t)B_ * 4);
    float* gh   = (float*)take((size_t)B_ * G_ * 4);
    float* h0   = (float*)take((size_t)B_ * H_ * 4);
    float* h1   = (float*)take((size_t)B_ * H_ * 4);
    unsigned short* hb0 = (unsigned short*)take((size_t)B_ * H_ * 2);
    unsigned short* hb1 = (unsigned short*)take((size_t)B_ * H_ * 2);

    float* out   = (float*)d_out;
    float* hlast = out + (size_t)B_ * S_ * H_;

    // 1) Convert operands to bf16 once per call
    cvt_bf16_kernel<<<4096, 256, 0, stream>>>(x,   xbf,  B_ * S_ * I_);
    cvt_bf16_kernel<<<1536, 256, 0, stream>>>(Wih, wihb, G_ * I_);
    cvt_bf16_kernel<<<3072, 256, 0, stream>>>(Whh, whhb, G_ * H_);

    // 2) Precompute x-part of attention scores
    sx_kernel<<<(B_ * S_) / 8, 256, 0, stream>>>(x, Watt, sx);

    // 3) h0 = 0 (fp32 + bf16)
    zero_h_kernel<<<512, 256, 0, stream>>>(h0, hb0, B_ * H_);

    // 4) Hoisted x-path GEMM: GI[B*S, 3H] = x[B*S, I] @ W_ih^T
    //    Fat wave tile (32x64) for the throughput-bound parallel GEMM.
    {
        dim3 grid((B_ * S_) / 64, G_ / 256);
        gemm_bf16_nt<4><<<grid, 256, 0, stream>>>(xbf, wihb, GI, B_ * S_, G_, I_);
    }

    // 5) Serial scan: softmax -> gh GEMM (WMMA) -> fused gate update.
    //    Narrow wave tile (32x32) keeps 384 waves in flight on the
    //    latency-critical recurrent GEMM (M is only 128).
    dim3 ggh(B_ / 64, G_ / 128);
    for (int t = 0; t < S_; ++t) {
        const float*          hp  = (t & 1) ? h1  : h0;
        float*                hn  = (t & 1) ? h0  : h1;
        const unsigned short* hbp = (t & 1) ? hb1 : hb0;
        unsigned short*       hbn = (t & 1) ? hb0 : hb1;

        att_kernel<<<1, 128, 0, stream>>>(hp, Watt, batt, sx, t, attb);
        gemm_bf16_nt<2><<<ggh, 256, 0, stream>>>(hbp, whhb, gh, B_, G_, H_);
        combine_kernel<<<(B_ * H_) / 256, 256, 0, stream>>>(
            GI, gh, attb, bih, bhh, hp, hn, hbn, out,
            (t == S_ - 1) ? hlast : nullptr, t);
    }
}
